// SSMEncoderAdapter_49452253446285
// MI455X (gfx1250) — matrix operand order repaired and checked by reading, not compiled, not run on hardware
//
#include <hip/hip_runtime.h>
#include <math.h>

// ---------------------------------------------------------------------------
// Mamba-style SSM encoder for MI455X (gfx1250, wave32).
// All projections run through v_wmma_f32_16x16x32_f16 (f16 in, f32 accum).
// Every GEMM has K % 32 == 0 and N % 64 == 0 (dt K padded 16->32, xproj N
// padded 48->64), and the grid covers N exactly, so the GEMM kernel has NO
// divergent or wave-level guards: EXEC is all-ones at every WMMA and the
// inner loop is pure b128 loads + 4 WMMAs.
// ---------------------------------------------------------------------------

typedef _Float16 half_t;
typedef __attribute__((ext_vector_type(16))) _Float16 v16h;
typedef __attribute__((ext_vector_type(8)))  _Float16 v8h;
typedef __attribute__((ext_vector_type(8)))  float    v8f;

#define NLc 2
#define DMc 256
#define Fc  128
#define DIc 512
#define DSc 16
#define DRc 16
#define Kc  4
#define Bc  4
#define Lc  2048
#define BLc (Bc * Lc)
#define DTKP 32   // dt projection K padded 16 -> 32
#define XPN  64   // xproj output N padded 48 -> 64
#define DBCW 64   // dbc row stride (padded)

// ---------------------------------------------------------------------------
// WMMA GEMM:  C[M,N] (f32) = act( A[M,K](f16) * W[N,K]^T(f16) + bias )
// Requirements (guaranteed by call sites): K % 32 == 0, N % 64 == 0,
// M % 16 == 0, grid covers N exactly. Each wave computes a 16x64 strip:
// the A fragment is loaded once per K-step and reused for 4 WMMAs.
// act: 0 = none, 1 = softplus.  accum: 1 -> C += result (residual fuse).
// ---------------------------------------------------------------------------
__global__ void __launch_bounds__(128)
wmma_gemm_kernel(const half_t* __restrict__ A,
                 const half_t* __restrict__ W,
                 const float* __restrict__ bias,
                 float* __restrict__ C,
                 int M, int N, int Kd, int act, int accum)
{
    const int lane   = threadIdx.x;                       // 0..31
    const int nbase  = (blockIdx.x * blockDim.y + threadIdx.y) * 64;
    const int m0     = blockIdx.y * 16;

    const int half16 = lane >> 4;                         // 0 or 1
    const int l15    = lane & 15;
    const int arow   = m0 + l15;                          // A row for this lane

    v8f c0 = {}, c1 = {}, c2 = {}, c3 = {};

    const half_t* aRow = A + (size_t)arow * Kd;
    const half_t* wRow = W + (size_t)(nbase + l15) * Kd;  // W row = out column

    for (int kk = 0; kk < Kd; kk += 32) {
        union { v16h v; v8h h[2]; } a;
        a.h[0] = *(const v8h*)(aRow + kk + half16 * 8);
        a.h[1] = *(const v8h*)(aRow + kk + half16 * 8 + 16);

        #pragma unroll
        for (int t = 0; t < 4; ++t) {
            const half_t* bp = wRow + (size_t)t * 16 * Kd + kk + half16 * 8;
            union { v16h v; v8h h[2]; } b;
            b.h[0] = *(const v8h*)(bp);
            b.h[1] = *(const v8h*)(bp + 16);
            v8f& acc = (t == 0) ? c0 : (t == 1) ? c1 : (t == 2) ? c2 : c3;
            acc = __builtin_amdgcn_wmma_f32_16x16x32_f16(
                      false, a.v, false, b.v, (short)0, acc, false, false);
        }
    }

    #pragma unroll
    for (int t = 0; t < 4; ++t) {
        const v8f& acc = (t == 0) ? c0 : (t == 1) ? c1 : (t == 2) ? c2 : c3;
        const int n = nbase + t * 16 + l15;
        const float bv = bias ? bias[n] : 0.f;
        #pragma unroll
        for (int g = 0; g < 8; ++g) {
            int m = m0 + g + half16 * 8;
            float v = acc[g] + bv;
            if (act == 1) v = (v > 20.f) ? v : log1pf(__expf(v));  // softplus
            size_t idx = (size_t)m * N + n;
            if (accum) C[idx] += v; else C[idx] = v;
        }
    }
}

// ---------------------------------------------------------------------------
__global__ void f32_to_f16_kernel(const float* __restrict__ src,
                                  half_t* __restrict__ dst, int n)
{
    int i = blockIdx.x * blockDim.x + threadIdx.x;
    if (i < n) dst[i] = (half_t)src[i];
}

// dt_w: rows of width ksrc (f32) -> rows of width kdst (f16), zero-padded cols
__global__ void f32_to_f16_pad_kernel(const float* __restrict__ src,
                                      half_t* __restrict__ dst,
                                      int rows, int ksrc, int kdst)
{
    int i = blockIdx.x * blockDim.x + threadIdx.x;
    if (i >= rows * kdst) return;
    int r = i / kdst, k = i % kdst;
    dst[i] = (k < ksrc) ? (half_t)src[(size_t)r * ksrc + k] : (half_t)0.f;
}

// xproj_w: [NL, 48, DI] f32 -> [NL, 64, DI] f16, rows 48..63 zeroed
__global__ void pad_xproj_w_kernel(const float* __restrict__ src,
                                   half_t* __restrict__ dst)
{
    int i = blockIdx.x * blockDim.x + threadIdx.x;
    if (i >= NLc * XPN * DIc) return;
    int layer = i / (XPN * DIc);
    int rem   = i % (XPN * DIc);
    int r = rem / DIc, k = rem % DIc;
    dst[i] = (r < DRc + 2 * DSc)
                 ? (half_t)src[((size_t)layer * (DRc + 2 * DSc) + r) * DIc + k]
                 : (half_t)0.f;
}

// rmsnorm over DM=256 per token row; outputs f16 (for next GEMM) and optional f32
__global__ void rmsnorm_kernel(const float* __restrict__ h,
                               const float* __restrict__ w,
                               half_t* __restrict__ out16,
                               float* __restrict__ out32)
{
    int row = blockIdx.x;
    int t   = threadIdx.x;                 // 0..255
    float v = h[(size_t)row * DMc + t];
    __shared__ float red[DMc];
    red[t] = v * v;
    __syncthreads();
    for (int s = DMc / 2; s > 0; s >>= 1) {
        if (t < s) red[t] += red[t + s];
        __syncthreads();
    }
    float scale = rsqrtf(red[0] / (float)DMc + 1e-6f);
    float o = v * scale * w[t];
    out16[(size_t)row * DMc + t] = (half_t)o;
    if (out32) out32[(size_t)row * DMc + t] = o;
}

// causal depthwise conv (K=4) over u = xz[:, :DI], + bias, SiLU; f32 + f16 out
__global__ void conv_silu_kernel(const float* __restrict__ xz,
                                 const float* __restrict__ cw,
                                 const float* __restrict__ cb,
                                 float* __restrict__ u32,
                                 half_t* __restrict__ u16)
{
    int gid = blockIdx.x * blockDim.x + threadIdx.x;
    if (gid >= BLc * DIc) return;
    int d  = gid % DIc;
    int bl = gid / DIc;
    int l  = bl % Lc;
    int b  = bl / Lc;
    float acc = cb[d];
    #pragma unroll
    for (int k = 0; k < Kc; ++k) {
        int lt = l - (Kc - 1) + k;
        if (lt >= 0)
            acc += cw[d * Kc + k] * xz[((size_t)(b * Lc + lt)) * (2 * DIc) + d];
    }
    float s = acc / (1.f + __expf(-acc));  // silu
    u32[gid] = s;
    u16[gid] = (half_t)s;
}

// extract dt = dbc[:, 0:16] as f16, zero-padded to K=32 per row
__global__ void extract_dt_kernel(const float* __restrict__ dbc,
                                  half_t* __restrict__ dt)
{
    int i = blockIdx.x * blockDim.x + threadIdx.x;
    if (i >= BLc * DTKP) return;
    int r = i / DTKP, j = i % DTKP;
    dt[i] = (j < DRc) ? (half_t)dbc[(size_t)r * DBCW + j] : (half_t)0.f;
}

// Selective scan: one (b,d) chain per lane, state hs[16] in registers.
// B/C vectors (shared across d within a (b,t)) staged through LDS, 64 steps
// per stage. Fuses y = (scan + u*D) * silu(z), emits f16 for out_proj.
__global__ void __launch_bounds__(DIc)
scan_kernel(const float* __restrict__ delta,
            const float* __restrict__ u,
            const float* __restrict__ dbc,
            const float* __restrict__ xz,
            const float* __restrict__ A_log,
            const float* __restrict__ Dw,
            half_t* __restrict__ yout)
{
    const int b = blockIdx.x;
    const int d = threadIdx.x;             // 0..511
    float A[DSc];
    #pragma unroll
    for (int n = 0; n < DSc; ++n) A[n] = -__expf(A_log[d * DSc + n]);
    const float Dd = Dw[d];
    float hs[DSc];
    #pragma unroll
    for (int n = 0; n < DSc; ++n) hs[n] = 0.f;

    __shared__ float sB[64][DSc];
    __shared__ float sC[64][DSc];

    for (int t0 = 0; t0 < Lc; t0 += 64) {
        for (int i = threadIdx.x; i < 64 * 32; i += DIc) {
            int tt = i >> 5, j = i & 31;
            float v = dbc[((size_t)(b * Lc + t0 + tt)) * DBCW + DRc + j];
            if (j < DSc) sB[tt][j] = v; else sC[tt][j - DSc] = v;
        }
        __syncthreads();
        for (int tt = 0; tt < 64; ++tt) {
            size_t row = (size_t)(b * Lc) + t0 + tt;
            float dl = delta[row * DIc + d];
            float uu = u[row * DIc + d];
            float du = dl * uu;
            float y = 0.f;
            #pragma unroll
            for (int n = 0; n < DSc; ++n) {
                float dA = __expf(dl * A[n]);
                hs[n] = dA * hs[n] + du * sB[tt][n];
                y += hs[n] * sC[tt][n];
            }
            float z  = xz[row * (2 * DIc) + DIc + d];
            float sz = z / (1.f + __expf(-z));
            yout[row * DIc + d] = (half_t)((y + uu * Dd) * sz);
        }
        __syncthreads();
    }
}

__global__ void pool_kernel(const float* __restrict__ hn, float* __restrict__ pooled)
{
    int b = blockIdx.x, dm = threadIdx.x;  // 256 threads
    float acc = 0.f;
    for (int t = 0; t < Lc; ++t) acc += hn[((size_t)(b * Lc) + t) * DMc + dm];
    pooled[b * DMc + dm] = acc / (float)Lc;
}

__global__ void proj_kernel(const float* __restrict__ pooled,
                            const float* __restrict__ pw,
                            const float* __restrict__ pb,
                            float* __restrict__ biasbuf)
{
    int i = blockIdx.x * blockDim.x + threadIdx.x;   // 4*128
    if (i >= Bc * Fc) return;
    int b = i / Fc, f = i % Fc;
    float acc = pb[f];
    for (int dm = 0; dm < DMc; ++dm) acc += pooled[b * DMc + dm] * pw[f * DMc + dm];
    biasbuf[i] = acc;
}

__global__ void final_add_kernel(const float* __restrict__ x,
                                 const float* __restrict__ biasbuf,
                                 float* __restrict__ out)
{
    int gid = blockIdx.x * blockDim.x + threadIdx.x;
    if (gid >= BLc * Fc) return;
    int f = gid % Fc;
    int b = gid / (Lc * Fc);
    out[gid] = x[gid] + biasbuf[b * Fc + f];
}

// ---------------------------------------------------------------------------
static void launch_gemm(const half_t* A, const half_t* W, const float* bias,
                        float* C, int M, int N, int Kd, int act, int accum,
                        hipStream_t s)
{
    const int strips = N / 64;                  // N % 64 == 0 guaranteed
    const int wy     = (strips >= 4) ? 4 : strips;
    dim3 block(32, wy, 1);
    dim3 grid(strips / wy, M / 16, 1);
    wmma_gemm_kernel<<<grid, block, 0, s>>>(A, W, bias, C, M, N, Kd, act, accum);
}

static void launch_cvt(const float* src, half_t* dst, int n, hipStream_t s)
{
    f32_to_f16_kernel<<<(n + 255) / 256, 256, 0, s>>>(src, dst, n);
}

extern "C" void kernel_launch(void* const* d_in, const int* in_sizes, int n_in,
                              void* d_out, int out_size, void* d_ws, size_t ws_size,
                              hipStream_t stream)
{
    (void)in_sizes; (void)n_in; (void)out_size; (void)ws_size;

    const float* x        = (const float*)d_in[0];
    const float* embed_w  = (const float*)d_in[1];
    const float* embed_b  = (const float*)d_in[2];
    const float* in_w     = (const float*)d_in[3];
    const float* conv_w   = (const float*)d_in[4];
    const float* conv_b   = (const float*)d_in[5];
    const float* xproj_w  = (const float*)d_in[6];
    const float* dt_w     = (const float*)d_in[7];
    const float* dt_b     = (const float*)d_in[8];
    const float* A_log    = (const float*)d_in[9];
    const float* Dp       = (const float*)d_in[10];
    const float* out_w    = (const float*)d_in[11];
    const float* norm_w   = (const float*)d_in[12];
    const float* fnorm_w  = (const float*)d_in[13];
    const float* proj_w   = (const float*)d_in[14];
    const float* proj_b   = (const float*)d_in[15];
    float* out = (float*)d_out;

    // workspace carve-up (256B aligned)
    char* ws = (char*)d_ws;
    size_t off = 0;
    auto carve = [&](size_t bytes) -> void* {
        void* p = ws + off;
        off = (off + bytes + 255) & ~(size_t)255;
        return p;
    };
    float*  h      = (float*) carve((size_t)BLc * DMc * 4);
    half_t* hn16   = (half_t*)carve((size_t)BLc * DMc * 2);
    float*  hn32   = (float*) carve((size_t)BLc * DMc * 4);
    float*  xz     = (float*) carve((size_t)BLc * 2 * DIc * 4);
    float*  u32    = (float*) carve((size_t)BLc * DIc * 4);
    half_t* u16    = (half_t*)carve((size_t)BLc * DIc * 2);
    float*  dbc    = (float*) carve((size_t)BLc * DBCW * 4);
    half_t* dt16   = (half_t*)carve((size_t)BLc * DTKP * 2);
    float*  delta  = (float*) carve((size_t)BLc * DIc * 4);
    half_t* y16    = (half_t*)carve((size_t)BLc * DIc * 2);
    half_t* x16    = (half_t*)carve((size_t)BLc * Fc * 2);
    half_t* ew16   = (half_t*)carve((size_t)DMc * Fc * 2);
    half_t* inw16  = (half_t*)carve((size_t)NLc * 2 * DIc * DMc * 2);
    half_t* xpw16  = (half_t*)carve((size_t)NLc * XPN * DIc * 2);
    half_t* dtw16  = (half_t*)carve((size_t)NLc * DIc * DTKP * 2);
    half_t* ow16   = (half_t*)carve((size_t)NLc * DMc * DIc * 2);
    float*  pooled = (float*) carve((size_t)Bc * DMc * 4);
    float*  bbuf   = (float*) carve((size_t)Bc * Fc * 4);

    // f16 copies of activations input + all GEMM weights
    launch_cvt(x,       x16,   BLc * Fc,              stream);
    launch_cvt(embed_w, ew16,  DMc * Fc,              stream);
    launch_cvt(in_w,    inw16, NLc * 2 * DIc * DMc,   stream);
    launch_cvt(out_w,   ow16,  NLc * DMc * DIc,       stream);
    {   // xproj_w: [NL,48,DI] -> [NL,64,DI] zero row padding
        int n = NLc * XPN * DIc;
        pad_xproj_w_kernel<<<(n + 255) / 256, 256, 0, stream>>>(xproj_w, xpw16);
    }
    {   // dt_w: [NL*DI, 16] -> [NL*DI, 32] zero col padding
        int n = NLc * DIc * DTKP;
        f32_to_f16_pad_kernel<<<(n + 255) / 256, 256, 0, stream>>>(
            dt_w, dtw16, NLc * DIc, DRc, DTKP);
    }

    // embedding:  h = x @ embed_w^T + embed_b
    launch_gemm(x16, ew16, embed_b, h, BLc, DMc, Fc, 0, 0, stream);

    for (int i = 0; i < NLc; ++i) {
        const half_t* inw_i = inw16 + (size_t)i * 2 * DIc * DMc;
        const half_t* xpw_i = xpw16 + (size_t)i * XPN * DIc;
        const half_t* dtw_i = dtw16 + (size_t)i * DIc * DTKP;
        const half_t* ow_i  = ow16  + (size_t)i * DMc * DIc;

        rmsnorm_kernel<<<BLc, DMc, 0, stream>>>(h, norm_w + (size_t)i * DMc,
                                                hn16, nullptr);
        launch_gemm(hn16, inw_i, nullptr, xz, BLc, 2 * DIc, DMc, 0, 0, stream);

        conv_silu_kernel<<<(BLc * DIc + 255) / 256, 256, 0, stream>>>(
            xz, conv_w + (size_t)i * DIc * Kc, conv_b + (size_t)i * DIc, u32, u16);

        // dbc[:, 0:64] = u @ xproj_w_padded^T   (cols 48..63 are zero-weight)
        launch_gemm(u16, xpw_i, nullptr, dbc, BLc, XPN, DIc, 0, 0, stream);

        extract_dt_kernel<<<(BLc * DTKP + 255) / 256, 256, 0, stream>>>(dbc, dt16);

        // delta = softplus(dt @ dt_w^T + dt_b)   (K padded to 32)
        launch_gemm(dt16, dtw_i, dt_b + (size_t)i * DIc, delta,
                    BLc, DIc, DTKP, 1, 0, stream);

        scan_kernel<<<Bc, DIc, 0, stream>>>(delta, u32, dbc, xz,
                                            A_log + (size_t)i * DIc * DSc,
                                            Dp + (size_t)i * DIc, y16);

        // residual fuse:  h += y @ out_w^T
        launch_gemm(y16, ow_i, nullptr, h, BLc, DMc, DIc, 0, 1, stream);
    }

    rmsnorm_kernel<<<BLc, DMc, 0, stream>>>(h, fnorm_w, hn16, hn32);
    pool_kernel<<<Bc, DMc, 0, stream>>>(hn32, pooled);
    proj_kernel<<<2, 256, 0, stream>>>(pooled, proj_w, proj_b, bbuf);
    final_add_kernel<<<(BLc * Fc + 255) / 256, 256, 0, stream>>>(x, bbuf, out);
}
